// CausalSelfAttention_36807869727464
// MI455X (gfx1250) — matrix operand order repaired and checked by reading, not compiled, not run on hardware
//
#include <hip/hip_runtime.h>
#include <hip/hip_bf16.h>

// Causal self-attention for MI455X (gfx1250, wave32, WMMA f16->f32).
// S=2048, B=2, D=1024, H=16, Hd=64.
// GEMMs: 8-wave blocks, async-to-LDS double-buffered weight tiles,
// 32x64 output per wave (8 WMMAs / 32-deep k-step).
// Attention: flash-style online softmax, 16 query rows per wave.

#define SEQ    2048
#define BATCH  2
#define DMODEL 1024
#define NHEADS 16
#define HEADD  64
#define MROWS  (SEQ * BATCH)   // 4096 rows of the (s*B+b, D) activation matrix
#define WT_LD  40              // padded LDS row (halves): 80B rows -> conflict-free

typedef __attribute__((ext_vector_type(16))) _Float16 v16h;
typedef __attribute__((ext_vector_type(8)))  float    v8f;
typedef __attribute__((ext_vector_type(8)))  int      v8i;

// ---------------------------------------------------------------------------
// WMMA wrapper: D = A(16x32 f16) * B(32x16 f16) + C(16x16 f32)
// ---------------------------------------------------------------------------
__device__ __forceinline__ v8f wmma_f16(v16h a, v16h b, v8f c) {
  return __builtin_amdgcn_wmma_f32_16x16x32_f16(
      /*neg_a=*/false, a, /*neg_b=*/false, b,
      /*c_mod=*/(short)0, c, /*reuse_a=*/false, /*reuse_b=*/false);
}

// A-fragment (16x32, f16) from row-major src (row stride ld halves).
// ISA layout: lane<16 -> M=lane, VGPR j holds K = (j/4)*16 + (lane/16)*8 + 2*(j%4), +1.
__device__ __forceinline__ v16h load_a16x32(const _Float16* src, int row0,
                                            int ld, int col0, int lane) {
  const int half = (lane >> 4) & 1;
  const int m    = lane & 15;
  union { v8i i; v16h h; } u;
  const _Float16* base = src + (long)(row0 + m) * ld + col0;
#pragma unroll
  for (int j = 0; j < 8; ++j) {
    const int k = ((j >> 2) << 4) + (half << 3) + ((j & 3) << 1);
    u.i[j] = *(const int*)(base + k);
  }
  return u.h;
}

// B-fragment (32x16, f16). Source is row-major over (n, k): B(k,n) = src[n*ld + k].
// ISA layout: lane%16 = column n; VGPR j holds K = (j/4)*16 + (lane/16)*8 + 2*(j%4), +1.
__device__ __forceinline__ v16h load_b32x16(const _Float16* src, int n0,
                                            int ld, int k0, int lane) {
  const int khalf = (lane >> 4) & 1;
  const int n     = lane & 15;
  union { v8i i; v16h h; } u;
  const _Float16* base = src + (long)(n0 + n) * ld + k0 + (khalf << 3);
#pragma unroll
  for (int j = 0; j < 8; ++j) u.i[j] = *(const int*)(base + ((j >> 2) << 4) + ((j & 3) << 1));
  return u.h;
}

// ---------------------------------------------------------------------------
// Async global->LDS stage of one 16-byte chunk (gfx1250 ASYNCcnt path).
// Generic shared-aperture addresses map to LDS by truncation (ISA 10.2),
// so the low 32 bits of a __shared__ pointer are a valid LDS byte address.
// ---------------------------------------------------------------------------
__device__ __forceinline__ void async_copy_b128(const _Float16* gsrc,
                                                _Float16* ldst) {
  unsigned lds_addr = (unsigned)(unsigned long long)ldst;
  asm volatile("global_load_async_to_lds_b128 %0, %1, off"
               :: "v"(lds_addr), "v"(gsrc)
               : "memory");
}
__device__ __forceinline__ void wait_async_le1() {
  asm volatile("s_wait_asynccnt 0x1" ::: "memory");
}
__device__ __forceinline__ void wait_async_le0() {
  asm volatile("s_wait_asynccnt 0x0" ::: "memory");
}

// ---------------------------------------------------------------------------
// f32 -> f16 conversion (grid-stride)
// ---------------------------------------------------------------------------
__global__ void cvt_f32_to_f16(const float* __restrict__ in,
                               _Float16* __restrict__ out, int n) {
  for (int i = blockIdx.x * blockDim.x + threadIdx.x; i < n;
       i += gridDim.x * blockDim.x)
    out[i] = (_Float16)in[i];
}

// ---------------------------------------------------------------------------
// QKV projection: Y = Xh @ W^T. 256 threads (8 waves) per block.
// Block tile 256(M) x 64(N); wave tile 32x64. Weight tile (64n x 32k) staged
// to LDS with double-buffered async copies, shared by all 8 waves.
// Q,K written row-major per (b,h): [bh][s][d]; V transposed: [bh][d][s].
// ---------------------------------------------------------------------------
__global__ __launch_bounds__(256) void qkv_gemm(
    const _Float16* __restrict__ xh,
    const _Float16* __restrict__ wq, const _Float16* __restrict__ wk,
    const _Float16* __restrict__ wv,
    _Float16* __restrict__ Qb, _Float16* __restrict__ Kb,
    _Float16* __restrict__ Vt) {
  const int tid  = threadIdx.x;
  const int lane = tid & 31;
  const int wave = tid >> 5;
  const int m0 = blockIdx.x * 256 + wave * 32;
  const int n0 = blockIdx.y * 64;
  const int z  = blockIdx.z;
  const _Float16* W = (z == 0) ? wq : (z == 1) ? wk : wv;

  __shared__ _Float16 wt[2][64 * WT_LD];

  // staging pattern: 256 threads, thread -> (n = tid/4, 16B chunk = tid%4)
  const int sn = tid >> 2, sc = tid & 3;
  const _Float16* grow = W + (long)(n0 + sn) * DMODEL + sc * 8;
  _Float16* lrow = &wt[0][0] + sn * WT_LD + sc * 8;

  v8f acc[2][4] = {};
  async_copy_b128(grow, lrow);                       // prologue: k-step 0

  const int NK = DMODEL / 32;
  for (int kk = 0; kk < NK; ++kk) {
    const int k0  = kk * 32;
    const int cur = kk & 1;
    if (kk + 1 < NK) {
      async_copy_b128(grow + (k0 + 32),
                      lrow + (cur ^ 1) * (64 * WT_LD));
      wait_async_le1();                              // k-step kk resident
    } else {
      wait_async_le0();
    }
    __syncthreads();

    const v16h a0 = load_a16x32(xh, m0,      DMODEL, k0, lane);
    const v16h a1 = load_a16x32(xh, m0 + 16, DMODEL, k0, lane);
    const _Float16* bt = &wt[cur][0];
#pragma unroll
    for (int c = 0; c < 4; ++c) {
      const v16h b = load_b32x16(bt, c * 16, WT_LD, 0, lane);
      acc[0][c] = wmma_f16(a0, b, acc[0][c]);
      acc[1][c] = wmma_f16(a1, b, acc[1][c]);
    }
    __syncthreads();                                 // done reading buf[cur]
  }

  // C layout: VGPR r -> rows r / r+8 (by half-wave), lane%16 -> column.
  const int half = lane >> 4, nl = lane & 15;
#pragma unroll
  for (int g = 0; g < 2; ++g) {
#pragma unroll
    for (int c = 0; c < 4; ++c) {
#pragma unroll
      for (int r = 0; r < 8; ++r) {
        const int m = m0 + g * 16 + r + 8 * half;  // flat row = s*B + b
        const int s = m >> 1, b = m & 1;           // BATCH == 2
        const int n = n0 + c * 16 + nl;
        const int h = n >> 6, d = n & 63;
        const _Float16 v = (_Float16)acc[g][c][r];
        const long bh = (long)(b * NHEADS + h);
        if (z == 0)      Qb[(bh * SEQ + s) * HEADD + d] = v;
        else if (z == 1) Kb[(bh * SEQ + s) * HEADD + d] = v;
        else             Vt[(bh * HEADD + d) * SEQ + s] = v;   // transposed
      }
    }
  }
}

// ---------------------------------------------------------------------------
// Flash-style causal attention. One wave handles 16 query rows of one (b,h).
// Per 32-key block: 4 WMMAs for S = Q@K^T, online softmax (f32 VALU + shfl
// row reductions), P re-laid-out C->A via LDS, 4 WMMAs for O += P@V.
// ---------------------------------------------------------------------------
__global__ __launch_bounds__(32) void attn_kernel(
    const _Float16* __restrict__ Qb, const _Float16* __restrict__ Kb,
    const _Float16* __restrict__ Vt, _Float16* __restrict__ attnh) {
  const int lane = threadIdx.x;
  const int qt = blockIdx.x;        // query tile (16 rows)
  const int bh = blockIdx.y;        // b*NHEADS + h
  const _Float16* Q = Qb + (long)bh * SEQ * HEADD;
  const _Float16* K = Kb + (long)bh * SEQ * HEADD;
  const _Float16* V = Vt + (long)bh * HEADD * SEQ;

  __shared__ _Float16 Pl[16 * 32];

  const v16h qa0 = load_a16x32(Q, qt * 16, HEADD, 0, lane);
  const v16h qa1 = load_a16x32(Q, qt * 16, HEADD, 32, lane);

  const int half = lane >> 4, nl = lane & 15;
  float rowm[8], rowl[8];
  v8f o[4] = {};
#pragma unroll
  for (int r = 0; r < 8; ++r) { rowm[r] = -1e30f; rowl[r] = 0.f; }

  const int nkt = ((qt * 16 + 15) >> 5);   // inclusive last 32-key block
  for (int kt = 0; kt <= nkt; ++kt) {
    const int t0 = kt * 32;
    // ---- scores: 16x32 tile, contraction over d (two k-steps of 32) ----
    v8f s0 = {}, s1 = {};
    s0 = wmma_f16(qa0, load_b32x16(K, t0,      HEADD, 0,  lane), s0);
    s0 = wmma_f16(qa1, load_b32x16(K, t0,      HEADD, 32, lane), s0);
    s1 = wmma_f16(qa0, load_b32x16(K, t0 + 16, HEADD, 0,  lane), s1);
    s1 = wmma_f16(qa1, load_b32x16(K, t0 + 16, HEADD, 32, lane), s1);

    // ---- online softmax (per row; rows live inside one 16-lane half) ----
#pragma unroll
    for (int r = 0; r < 8; ++r) {
      const int sq = qt * 16 + r + 8 * half;                 // query position
      float x0 = (t0 + nl      <= sq) ? s0[r] * 0.125f : -1e30f;
      float x1 = (t0 + 16 + nl <= sq) ? s1[r] * 0.125f : -1e30f;
      float bm = fmaxf(x0, x1);
      bm = fmaxf(bm, __shfl_xor(bm, 1));
      bm = fmaxf(bm, __shfl_xor(bm, 2));
      bm = fmaxf(bm, __shfl_xor(bm, 4));
      bm = fmaxf(bm, __shfl_xor(bm, 8));
      const float mnew = fmaxf(rowm[r], bm);
      const float p0 = __expf(x0 - mnew);
      const float p1 = __expf(x1 - mnew);
      const float sc = __expf(rowm[r] - mnew);
      float ps = p0 + p1;
      ps += __shfl_xor(ps, 1);
      ps += __shfl_xor(ps, 2);
      ps += __shfl_xor(ps, 4);
      ps += __shfl_xor(ps, 8);
      rowl[r] = rowl[r] * sc + ps;
      rowm[r] = mnew;
#pragma unroll
      for (int c = 0; c < 4; ++c) o[c][r] *= sc;
      const int m = r + 8 * half;
      Pl[m * 32 + nl]      = (_Float16)p0;
      Pl[m * 32 + 16 + nl] = (_Float16)p1;
    }
    __syncthreads();

    // ---- O += P(16x32) @ V(32x64) : 4 WMMAs, V is pre-transposed [d][t] ----
    const v16h pa = load_a16x32(Pl, 0, 32, 0, lane);
#pragma unroll
    for (int c = 0; c < 4; ++c) {
      v16h vb = load_b32x16(V, c * 16, SEQ, t0, lane);
      o[c] = wmma_f16(pa, vb, o[c]);
    }
    __syncthreads();
  }

  // ---- normalize and store as f16 rows of the (s*B+b, D) activation ----
  const int h = bh & (NHEADS - 1), b = bh >> 4;
#pragma unroll
  for (int c = 0; c < 4; ++c) {
#pragma unroll
    for (int r = 0; r < 8; ++r) {
      const int m = r + 8 * half;
      const int s = qt * 16 + m;
      const int col = h * HEADD + c * 16 + nl;
      attnh[((long)(s * BATCH + b)) * DMODEL + col] =
          (_Float16)(o[c][r] / rowl[r]);
    }
  }
}

// ---------------------------------------------------------------------------
// Output projection: out = attnh @ Wo^T (f32 result to d_out). Same staged
// structure as qkv_gemm.
// ---------------------------------------------------------------------------
__global__ __launch_bounds__(256) void out_gemm(
    const _Float16* __restrict__ ah, const _Float16* __restrict__ wo,
    float* __restrict__ out) {
  const int tid  = threadIdx.x;
  const int lane = tid & 31;
  const int wave = tid >> 5;
  const int m0 = blockIdx.x * 256 + wave * 32;
  const int n0 = blockIdx.y * 64;

  __shared__ _Float16 wt[2][64 * WT_LD];

  const int sn = tid >> 2, sc = tid & 3;
  const _Float16* grow = wo + (long)(n0 + sn) * DMODEL + sc * 8;
  _Float16* lrow = &wt[0][0] + sn * WT_LD + sc * 8;

  v8f acc[2][4] = {};
  async_copy_b128(grow, lrow);

  const int NK = DMODEL / 32;
  for (int kk = 0; kk < NK; ++kk) {
    const int k0  = kk * 32;
    const int cur = kk & 1;
    if (kk + 1 < NK) {
      async_copy_b128(grow + (k0 + 32), lrow + (cur ^ 1) * (64 * WT_LD));
      wait_async_le1();
    } else {
      wait_async_le0();
    }
    __syncthreads();

    const v16h a0 = load_a16x32(ah, m0,      DMODEL, k0, lane);
    const v16h a1 = load_a16x32(ah, m0 + 16, DMODEL, k0, lane);
    const _Float16* bt = &wt[cur][0];
#pragma unroll
    for (int c = 0; c < 4; ++c) {
      const v16h b = load_b32x16(bt, c * 16, WT_LD, 0, lane);
      acc[0][c] = wmma_f16(a0, b, acc[0][c]);
      acc[1][c] = wmma_f16(a1, b, acc[1][c]);
    }
    __syncthreads();
  }

  const int half = lane >> 4, nl = lane & 15;
#pragma unroll
  for (int g = 0; g < 2; ++g)
#pragma unroll
    for (int c = 0; c < 4; ++c)
#pragma unroll
      for (int r = 0; r < 8; ++r)
        out[(long)(m0 + g * 16 + r + 8 * half) * DMODEL + n0 + c * 16 + nl] =
            acc[g][c][r];
}

// ---------------------------------------------------------------------------
extern "C" void kernel_launch(void* const* d_in, const int* in_sizes, int n_in,
                              void* d_out, int out_size, void* d_ws,
                              size_t ws_size, hipStream_t stream) {
  (void)in_sizes; (void)n_in; (void)out_size; (void)ws_size;
  const float* x  = (const float*)d_in[0];
  const float* Wq = (const float*)d_in[1];
  const float* Wk = (const float*)d_in[2];
  const float* Wv = (const float*)d_in[3];
  const float* Wo = (const float*)d_in[4];
  float* out = (float*)d_out;

  // workspace layout (all f16), 48 MB total
  char* ws = (char*)d_ws;
  const size_t SZ_X = (size_t)MROWS * DMODEL * 2;                // 8 MB
  const size_t SZ_W = (size_t)DMODEL * DMODEL * 2;               // 2 MB
  const size_t SZ_Q = (size_t)BATCH * NHEADS * SEQ * HEADD * 2;  // 8 MB
  _Float16* xh  = (_Float16*)(ws);
  _Float16* wqh = (_Float16*)(ws + SZ_X);
  _Float16* wkh = (_Float16*)(ws + SZ_X + SZ_W);
  _Float16* wvh = (_Float16*)(ws + SZ_X + 2 * SZ_W);
  _Float16* woh = (_Float16*)(ws + SZ_X + 3 * SZ_W);
  _Float16* Qb  = (_Float16*)(ws + SZ_X + 4 * SZ_W);
  _Float16* Kb  = (_Float16*)(ws + SZ_X + 4 * SZ_W + SZ_Q);
  _Float16* Vt  = (_Float16*)(ws + SZ_X + 4 * SZ_W + 2 * SZ_Q);
  _Float16* ah  = (_Float16*)(ws + SZ_X + 4 * SZ_W + 3 * SZ_Q);

  const int NX = MROWS * DMODEL;      // 4 Mi elems
  const int NW = DMODEL * DMODEL;     // 1 Mi elems
  cvt_f32_to_f16<<<2048, 256, 0, stream>>>(x, xh, NX);
  cvt_f32_to_f16<<<1024, 256, 0, stream>>>(Wq, wqh, NW);
  cvt_f32_to_f16<<<1024, 256, 0, stream>>>(Wk, wkh, NW);
  cvt_f32_to_f16<<<1024, 256, 0, stream>>>(Wv, wvh, NW);
  cvt_f32_to_f16<<<1024, 256, 0, stream>>>(Wo, woh, NW);

  qkv_gemm<<<dim3(MROWS / 256, DMODEL / 64, 3), 256, 0, stream>>>(
      xh, wqh, wkh, wvh, Qb, Kb, Vt);

  attn_kernel<<<dim3(SEQ / 16, BATCH * NHEADS), 32, 0, stream>>>(Qb, Kb, Vt, ah);

  out_gemm<<<dim3(MROWS / 256, DMODEL / 64), 256, 0, stream>>>(ah, woh, out);
}
